// SelfAttention_64415919505520
// MI455X (gfx1250) — matrix operand order repaired and checked by reading, not compiled, not run on hardware
//
#include <hip/hip_runtime.h>

// Self-attention forward for B=1, T=4096, D=1024, H=16, HD=64 on gfx1250.
// Pipeline: f32->f16 convert, QKV GEMM (f16 WMMA, f32 accum, TDM-staged B),
// RoPE in-place, causal flash-attention (WMMA scores + WMMA P*V),
// output GEMM to f32.

typedef __attribute__((ext_vector_type(16))) _Float16     v16h;
typedef __attribute__((ext_vector_type(8)))  float        v8f;
typedef __attribute__((ext_vector_type(8)))  unsigned int v8u;
typedef __attribute__((ext_vector_type(4)))  unsigned int v4u;
typedef __attribute__((ext_vector_type(8)))  int          v8i;
typedef __attribute__((ext_vector_type(4)))  int          v4i;

#define T_SEQ   4096
#define D_MODEL 1024
#define N_HEADS 16
#define HDIM    64
#define QKV_LD  (3 * D_MODEL)

__device__ __forceinline__ v8f wmma16x16x32(v16h a, v16h b, v8f c) {
  // emits v_wmma_f32_16x16x32_f16
  return __builtin_amdgcn_wmma_f32_16x16x32_f16(false, a, false, b, (short)0, c,
                                                false, false);
}

// ---- WMMA fragment loaders (contiguous K order k = 16*(lane>>4) + s, wave32) ----------

// A-matrix 16x32 f16 (MxK). Lane l: row=l&15, VGPR j holds K = g*16+2j, 2j+1
// with g = lane>>4. Consecutive K pairs -> b32 loads.
__device__ __forceinline__ v16h load_a_frag(const _Float16* __restrict__ base,
                                            int ld, int row0, int k0, int lane) {
  int r = lane & 15, g = lane >> 4;
  const _Float16* p = base + (size_t)(row0 + r) * ld + k0 + g * 16;
  v8u u;
#pragma unroll
  for (int j = 0; j < 8; ++j) u[j] = *(const unsigned int*)(p + 2 * j);
  return __builtin_bit_cast(v16h, u);
}

// B-matrix 32x16 f16 where logical B[k][n] = Src[col0+n][k0+k] (Src row-major
// with leading dim ld) -- i.e. B = Src^T restricted to a tile. VGPR j holds
// K = g*16 + 2j, 2j+1 at column lane&15, which is contiguous in Src's row.
__device__ __forceinline__ v16h load_bt_frag(const _Float16* __restrict__ base,
                                             int ld, int col0, int k0, int lane) {
  int c = lane & 15, g = lane >> 4;
  const _Float16* p = base + (size_t)(col0 + c) * ld + k0 + g * 16;
  v8u u;
#pragma unroll
  for (int j = 0; j < 8; ++j) u[j] = *(const unsigned int*)(p + 2 * j);
  return __builtin_bit_cast(v16h, u);
}

// B-matrix 32x16 f16 where logical B[k][n] = Src[k0+k][col0+n] (natural
// row-major rows = K). K pairs live in different rows -> two 16-bit loads.
__device__ __forceinline__ v16h load_b_frag(const _Float16* __restrict__ base,
                                            int ld, int k0, int col0, int lane) {
  int c = lane & 15, g = lane >> 4;
  v16h b;
#pragma unroll
  for (int j = 0; j < 8; ++j) {
    b[2 * j]     = base[(size_t)(k0 + g * 16 + 2 * j) * ld + col0 + c];
    b[2 * j + 1] = base[(size_t)(k0 + g * 16 + 2 * j + 1) * ld + col0 + c];
  }
  return b;
}

// ---- Tensor Data Mover: stage a 2-D f16 tile into LDS ----------------------
// Builds a D# per CDNA5 ISA ch.8: group0 {count, lds_addr, global_addr, type=2},
// group1 {data_size=2B, tensor dims, tile dims, dim0 stride}, groups 2/3 zero
// (2-D tile, iterate/gather/multicast disabled). This toolchain exposes the
// 6-argument builtin: (u32x4, i32x8, i32x4, i32x4, i32x8, i32 cpol).

#if __has_builtin(__builtin_amdgcn_tensor_load_to_lds)
#define HAVE_TDM 1
__device__ __forceinline__ void tdm_load_tile_f16(void* lds_dst, const void* gsrc,
                                                  unsigned tensor_d0,   // elems
                                                  unsigned tensor_d1,   // rows
                                                  unsigned tile_d0,     // elems
                                                  unsigned tile_d1,     // rows
                                                  unsigned stride0) {   // elems
  unsigned long long ga = (unsigned long long)(size_t)gsrc;
  unsigned lds_off = (unsigned)(size_t)lds_dst;  // generic LDS addr low 32 bits
  v4u g0;
  g0[0] = 1u;                                         // count=1, user mode
  g0[1] = lds_off;                                    // lds_addr
  g0[2] = (unsigned)(ga & 0xFFFFFFFFu);               // global_addr[31:0]
  g0[3] = (unsigned)((ga >> 32) & 0x01FFFFFFu)        // global_addr[56:32]
          | (2u << 30);                               // type = 2 ("image")
  v8i g1;
  g1[0] = (int)(1u << 16);                            // data_size=1 -> 2 bytes
  g1[1] = (int)((tensor_d0 & 0xFFFFu) << 16);         // tensor_dim0[15:0]
  g1[2] = (int)((tensor_d0 >> 16) |
                ((tensor_d1 & 0xFFFFu) << 16));       // dim0[31:16], dim1[15:0]
  g1[3] = (int)((tensor_d1 >> 16) |
                ((tile_d0 & 0xFFFFu) << 16));         // dim1[31:16], tile_dim0
  g1[4] = (int)(tile_d1 & 0xFFFFu);                   // tile_dim1; tile_dim2=0
  g1[5] = (int)stride0;                               // dim0_stride[31:0]
  g1[6] = 0;                                          // stride0 hi, stride1 lo
  g1[7] = 0;
  v4i z4 = {0, 0, 0, 0};
  v8i z8 = {0, 0, 0, 0, 0, 0, 0, 0};
  __builtin_amdgcn_tensor_load_to_lds(g0, g1, z4, z4, z8, 0);
}
#else
#define HAVE_TDM 0
#endif

// ---- f32 -> f16 conversion of x, W_qkv, W_proj ------------------------------

__global__ void convert_f32_to_f16(const float* __restrict__ x,
                                   const float* __restrict__ wqkv,
                                   const float* __restrict__ wproj,
                                   _Float16* __restrict__ xh,
                                   _Float16* __restrict__ wqkvh,
                                   _Float16* __restrict__ wprojh) {
  size_t i = (size_t)blockIdx.x * blockDim.x + threadIdx.x;
  const size_t nx = (size_t)T_SEQ * D_MODEL;
  const size_t nq = (size_t)3 * D_MODEL * D_MODEL;
  const size_t np = (size_t)D_MODEL * D_MODEL;
  if (i < nx)                xh[i] = (_Float16)x[i];
  else if (i < nx + nq)      wqkvh[i - nx] = (_Float16)wqkv[i - nx];
  else if (i < nx + nq + np) wprojh[i - nx - nq] = (_Float16)wproj[i - nx - nq];
}

// ---- GEMM: C[M,N] = A[M,K] * B[N,K]^T  (torch-Linear convention) -----------
// 4 waves/block; block computes 64x64 of C, each wave a 16x64 strip.
// The shared 64(N)x32(K) B tile is DMA'd into LDS once per K-step by the TDM
// (wave 0 issues under a scalar branch since TDM ignores EXEC), then all four
// waves read their B fragments from LDS.

template <bool OUT_F16>
__global__ void __launch_bounds__(128)
gemm_nt(const _Float16* __restrict__ A, const _Float16* __restrict__ B,
        _Float16* __restrict__ Ch, float* __restrict__ Cf,
        int M, int N, int K) {
  __shared__ _Float16 Bs[64][32];
  int lane = threadIdx.x & 31;
  int wave = threadIdx.x >> 5;
  int m0 = blockIdx.y * 64 + wave * 16;
  int n0 = blockIdx.x * 64;
  (void)M;

  v8f acc[4] = {};
  for (int k0 = 0; k0 < K; k0 += 32) {
#if HAVE_TDM
    if (__builtin_amdgcn_readfirstlane(wave) == 0) {
      tdm_load_tile_f16(&Bs[0][0], B + (size_t)n0 * K + k0,
                        (unsigned)K, (unsigned)N, 32u, 64u, (unsigned)K);
      __builtin_amdgcn_s_wait_tensorcnt(0);
    }
#else
    for (int i = threadIdx.x; i < 64 * 32; i += 128)
      (&Bs[0][0])[i] = B[(size_t)(n0 + (i >> 5)) * K + k0 + (i & 31)];
#endif
    __syncthreads();

    v16h a = load_a_frag(A, K, m0, k0, lane);
    if (k0 + 32 < K)
      __builtin_prefetch(A + (size_t)(m0 + (lane & 15)) * K + k0 + 32, 0, 3);
#pragma unroll
    for (int t = 0; t < 4; ++t) {
      v16h b = load_bt_frag(&Bs[0][0], 32, t * 16, 0, lane);
      acc[t] = wmma16x16x32(a, b, acc[t]);
    }
    __syncthreads();
  }

  int c = lane & 15, g = lane >> 4;
#pragma unroll
  for (int t = 0; t < 4; ++t) {
#pragma unroll
    for (int r = 0; r < 8; ++r) {
      size_t idx = (size_t)(m0 + g * 8 + r) * N + n0 + t * 16 + c;
      if (OUT_F16) Ch[idx] = (_Float16)acc[t][r];
      else         Cf[idx] = acc[t][r];
    }
  }
}

// ---- RoPE applied in place to the q and k slices of qkv[T, 3D] -------------
// rotate_half([x1,x2]) = [-x2,x1]; pair (i, i+32), angle = t * base^(-i/32).

__global__ void rope_inplace(_Float16* __restrict__ qkv) {
  size_t i = (size_t)blockIdx.x * blockDim.x + threadIdx.x;  // T*H*32 threads
  int pair = (int)(i & 31);
  int h    = (int)((i >> 5) & (N_HEADS - 1));
  int t    = (int)(i >> 9);
  if (t >= T_SEQ) return;
  float inv = __powf(10000.0f, -(float)pair / 32.0f);
  float s, co;
  __sincosf((float)t * inv, &s, &co);
  size_t bq = (size_t)t * QKV_LD + h * HDIM + pair;
  size_t bk = bq + D_MODEL;
#pragma unroll
  for (int which = 0; which < 2; ++which) {
    size_t b = which ? bk : bq;
    float x1 = (float)qkv[b];
    float x2 = (float)qkv[b + 32];
    qkv[b]      = (_Float16)(x1 * co - x2 * s);
    qkv[b + 32] = (_Float16)(x2 * co + x1 * s);
  }
}

// ---- Causal flash attention: one wave per (head, 16-query tile) ------------
// Scores: S(16x16) = q(16x64) * K^T via two chained K=32 WMMAs.
// Online softmax in f32; P goes D-layout -> LDS -> A-layout; out += P*V via
// one K=32 WMMA per 16-wide d-chunk.

__global__ void __launch_bounds__(128)
flash_attn(const _Float16* __restrict__ qkv, _Float16* __restrict__ attout) {
  __shared__ _Float16 Pt[4][16][32];
  int lane = threadIdx.x & 31;
  int wave = threadIdx.x >> 5;
  int h  = blockIdx.y;
  int q0 = (blockIdx.x * 4 + wave) * 16;
  int c = lane & 15, g = lane >> 4;

  const _Float16* qbase = qkv + h * HDIM;
  const _Float16* kbase = qkv + D_MODEL + h * HDIM;
  const _Float16* vbase = qkv + 2 * D_MODEL + h * HDIM;

  v16h aq0 = load_a_frag(qbase, QKV_LD, q0, 0, lane);
  v16h aq1 = load_a_frag(qbase, QKV_LD, q0, 32, lane);

  v8f o[4] = {};
  float mrow[8], lrow[8];
#pragma unroll
  for (int r = 0; r < 8; ++r) { mrow[r] = -3.0e38f; lrow[r] = 0.0f; }

  for (int kb = 0; kb < q0 + 16; kb += 32) {
#pragma unroll
    for (int s = 0; s < 2; ++s) {
      v16h bk0 = load_bt_frag(kbase, QKV_LD, kb + s * 16, 0, lane);
      v16h bk1 = load_bt_frag(kbase, QKV_LD, kb + s * 16, 32, lane);
      v8f S = {};
      S = wmma16x16x32(aq0, bk0, S);
      S = wmma16x16x32(aq1, bk1, S);
#pragma unroll
      for (int r = 0; r < 8; ++r) {
        int query = q0 + g * 8 + r;      // D-layout row
        int key   = kb + s * 16 + c;     // D-layout col
        float sv = S[r] * 0.125f;        // 1/sqrt(64)
        if (key > query) sv = -3.0e38f;  // causal mask
        // row-max across the 16 lanes holding this row
        float mx = sv;
        mx = fmaxf(mx, __shfl_xor(mx, 1));
        mx = fmaxf(mx, __shfl_xor(mx, 2));
        mx = fmaxf(mx, __shfl_xor(mx, 4));
        mx = fmaxf(mx, __shfl_xor(mx, 8));
        float mnew = fmaxf(mrow[r], mx);
        float p    = __expf(sv - mnew);
        float corr = __expf(mrow[r] - mnew);
        float ps = p;
        ps += __shfl_xor(ps, 1);
        ps += __shfl_xor(ps, 2);
        ps += __shfl_xor(ps, 4);
        ps += __shfl_xor(ps, 8);
        lrow[r] = lrow[r] * corr + ps;
        mrow[r] = mnew;
#pragma unroll
        for (int t = 0; t < 4; ++t) o[t][r] *= corr;
        Pt[wave][g * 8 + r][s * 16 + c] = (_Float16)p;
      }
    }
    // make the wave's LDS stores visible before A-layout reload
    asm volatile("s_wait_dscnt 0" ::: "memory");
    v16h pa = load_a_frag(&Pt[wave][0][0], 32, 0, 0, lane);
#pragma unroll
    for (int t = 0; t < 4; ++t) {
      v16h bv = load_b_frag(vbase, QKV_LD, kb, t * 16, lane);
      o[t] = wmma16x16x32(pa, bv, o[t]);
    }
  }

#pragma unroll
  for (int t = 0; t < 4; ++t) {
#pragma unroll
    for (int r = 0; r < 8; ++r) {
      size_t idx = (size_t)(q0 + g * 8 + r) * D_MODEL + h * HDIM + t * 16 + c;
      attout[idx] = (_Float16)(o[t][r] / lrow[r]);
    }
  }
}

// ---- driver -----------------------------------------------------------------

extern "C" void kernel_launch(void* const* d_in, const int* in_sizes, int n_in,
                              void* d_out, int out_size, void* d_ws, size_t ws_size,
                              hipStream_t stream) {
  (void)in_sizes; (void)n_in; (void)out_size; (void)ws_size;
  const float* x     = (const float*)d_in[0];
  const float* wqkv  = (const float*)d_in[1];
  const float* wproj = (const float*)d_in[2];
  float* out = (float*)d_out;

  char* ws = (char*)d_ws;
  size_t off = 0;
  _Float16* xh     = (_Float16*)(ws + off); off += (size_t)T_SEQ * D_MODEL * 2;        // 8 MB
  _Float16* wqkvh  = (_Float16*)(ws + off); off += (size_t)3 * D_MODEL * D_MODEL * 2;  // 6 MB
  _Float16* wprojh = (_Float16*)(ws + off); off += (size_t)D_MODEL * D_MODEL * 2;      // 2 MB
  _Float16* qkvh   = (_Float16*)(ws + off); off += (size_t)T_SEQ * 3 * D_MODEL * 2;    // 24 MB
  _Float16* attout = (_Float16*)(ws + off); off += (size_t)T_SEQ * D_MODEL * 2;        // 8 MB

  // 1) f32 -> f16 of all operands
  size_t ncvt = (size_t)T_SEQ * D_MODEL + (size_t)3 * D_MODEL * D_MODEL +
                (size_t)D_MODEL * D_MODEL;  // 8388608
  convert_f32_to_f16<<<(unsigned)((ncvt + 255) / 256), 256, 0, stream>>>(
      x, wqkv, wproj, xh, wqkvh, wprojh);

  // 2) qkv[T,3D] = x[T,D] * W_qkv[3D,D]^T
  gemm_nt<true><<<dim3(3 * D_MODEL / 64, T_SEQ / 64), 128, 0, stream>>>(
      xh, wqkvh, qkvh, nullptr, T_SEQ, 3 * D_MODEL, D_MODEL);

  // 3) RoPE on q and k slices
  size_t nrope = (size_t)T_SEQ * N_HEADS * 32;
  rope_inplace<<<(unsigned)(nrope / 256), 256, 0, stream>>>(qkvh);

  // 4) causal flash attention -> attout[T,D]
  flash_attn<<<dim3(T_SEQ / 64, N_HEADS), 128, 0, stream>>>(qkvh, attout);

  // 5) out[T,D] = attout[T,D] * W_proj[D,D]^T  (f32 result)
  gemm_nt<false><<<dim3(D_MODEL / 64, T_SEQ / 64), 128, 0, stream>>>(
      attout, wprojh, nullptr, out, T_SEQ, D_MODEL, D_MODEL);
}
